// att_57543971832338
// MI455X (gfx1250) — matrix operand order repaired and checked by reading, not compiled, not run on hardware
//
#include <hip/hip_runtime.h>
#include <hip/hip_bf16.h>

#define BB   32
#define CC   256
#define LL   1024
#define JB   32          // j-block width
#define KP   36          // padded LDS row stride (floats) for K/V/SD tiles (16B aligned rows)
#define PSTR 36          // per-wave P scratch row stride (floats)

typedef __attribute__((ext_vector_type(16))) __bf16 v16bf;
typedef __attribute__((ext_vector_type(8)))  __bf16 v8bf;
typedef __attribute__((ext_vector_type(8)))  float  v8f;
typedef __attribute__((ext_vector_type(4)))  float  f32x4;
typedef __attribute__((ext_vector_type(4)))  unsigned int u32x4;
typedef __attribute__((ext_vector_type(8)))  int    i32x8;
typedef __attribute__((ext_vector_type(4)))  int    i32x4;

// f32 -> bf16 via native converts (v_cvt_pk_bf16_f32, RNE)
static __device__ __forceinline__ __bf16 f2bf(float f) { return (__bf16)f; }

// reductions within each 16-lane half of the wave32
static __device__ __forceinline__ float redmax16(float x) {
  x = fmaxf(x, __shfl_xor(x, 1, 32));
  x = fmaxf(x, __shfl_xor(x, 2, 32));
  x = fmaxf(x, __shfl_xor(x, 4, 32));
  x = fmaxf(x, __shfl_xor(x, 8, 32));
  return x;
}
static __device__ __forceinline__ float redsum16(float x) {
  x += __shfl_xor(x, 1, 32);
  x += __shfl_xor(x, 2, 32);
  x += __shfl_xor(x, 4, 32);
  x += __shfl_xor(x, 8, 32);
  return x;
}

// TDM: DMA a [CC x JB] f32 tile (global row stride LL floats) into LDS,
// padding each 32-DWORD row with 4 DWORDs => LDS row stride KP=36 floats.
static __device__ __forceinline__ void tdm_load_tile(unsigned lds_off, const float* gsrc) {
  unsigned long long ga = (unsigned long long)(size_t)gsrc;
  u32x4 g0;
  g0[0] = 1u;                                                   // count=1, user flags 0
  g0[1] = lds_off;                                              // lds_addr
  g0[2] = (unsigned)(ga & 0xffffffffull);                       // global_addr[31:0]
  g0[3] = (unsigned)((ga >> 32) & 0x01ffffffull) | (2u << 30);  // addr[56:32] | type=2
  i32x8 g1;
  g1[0] = (int)((2u << 16) | (1u << 20) | (4u << 22) | (3u << 25)); // 4B elems, pad_en, interval=32dw, amount=4dw
  g1[1] = (int)(((unsigned)LL & 0xffffu) << 16);        // tensor_dim0 = L      (bits 79:48)
  g1[2] = (int)(((unsigned)CC & 0xffffu) << 16);        // tensor_dim1 = C      (bits 111:80)
  g1[3] = (int)(((unsigned)JB & 0xffffu) << 16);        // tile_dim0  = 32      (bits 127:112)
  g1[4] = (int)CC;                                      // tile_dim1  = 256     (bits 143:128)
  g1[5] = (int)LL;                                      // tensor_dim0_stride   (bits 207:160)
  g1[6] = 0;
  g1[7] = 0;
  i32x4 gz4; gz4[0] = 0; gz4[1] = 0; gz4[2] = 0; gz4[3] = 0;
  i32x8 gz8;
#pragma unroll
  for (int i = 0; i < 8; ++i) gz8[i] = 0;
  __builtin_amdgcn_tensor_load_to_lds(g0, g1, gz4, gz4, gz8, 0);
}

__global__ __launch_bounds__(128, 2)
void attn_fa(const float* __restrict__ q, const float* __restrict__ k,
             const float* __restrict__ v, const float* __restrict__ sd,
             const float* __restrict__ btab, const int* __restrict__ ridx,
             float* __restrict__ out) {
  const int tid  = threadIdx.x;
  const int wave = tid >> 5;
  const int lane = tid & 31;
  const int lh   = lane >> 4;   // lane half
  const int m    = lane & 15;

  const int b  = blockIdx.x >> 4;                        // batch
  const int i0 = ((blockIdx.x & 15) << 6) + (wave << 4); // this wave's 16-query tile

  extern __shared__ float smem[];
  float* Kt = smem;                // CC*KP  K tile (f32, padded rows)
  float* Vt = Kt + CC * KP;        // CC*KP  V tile
  float* Dt = Vt + CC * KP;        // CC*KP  std tile
  float* WsBase = Dt + CC * KP;    // 4 x (16*PSTR) per-wave P scratch
  float* Ws = WsBase + wave * (16 * PSTR);
  __bf16* Qw = (__bf16*)(WsBase + 4 * (16 * PSTR)) + (size_t)wave * 4096; // per-wave Q frags (bf16)

  const float* qb  = q  + (size_t)b * CC * LL;
  const float* kb_ = k  + (size_t)b * CC * LL;
  const float* vb  = v  + (size_t)b * CC * LL;
  const float* db  = sd + (size_t)b * CC * LL;

  // ---- load Q, convert to bf16, park the 8 A-fragments (16x32 each) in wave-private LDS.
  //      A layout: lane holds row M=m, element e -> K = (e&7) + 16*(e>>3) + 8*lh
#pragma unroll
  for (int kbk = 0; kbk < 8; ++kbk) {
    v16bf qfrag;
#pragma unroll
    for (int e = 0; e < 16; ++e) {
      int c = kbk * 32 + (e & 7) + ((e >> 3) << 4) + (lh << 3);
      qfrag[e] = f2bf(qb[c * LL + i0 + m]);
    }
    __bf16* dst = Qw + (kbk * 32 + lane) * 16;   // 32B per lane, conflict-free
    *(v8bf*)dst       = __builtin_shufflevector(qfrag, qfrag, 0, 1, 2, 3, 4, 5, 6, 7);
    *(v8bf*)(dst + 8) = __builtin_shufflevector(qfrag, qfrag, 8, 9, 10, 11, 12, 13, 14, 15);
  }

  // ---- flash state: C-layout row = r + 8*lh, col = m
  v8f accV[16], accD[16];
#pragma unroll
  for (int ct = 0; ct < 16; ++ct)
#pragma unroll
    for (int r = 0; r < 8; ++r) { accV[ct][r] = 0.f; accD[ct][r] = 0.f; }
  float mrow[8], lrow[8];
#pragma unroll
  for (int r = 0; r < 8; ++r) { mrow[r] = -1e30f; lrow[r] = 0.f; }

  for (int j0 = 0; j0 < LL; j0 += JB) {
    __syncthreads();                       // previous tiles fully consumed
    if (wave == 0) {
      tdm_load_tile((unsigned)(size_t)Kt, kb_ + j0);
      tdm_load_tile((unsigned)(size_t)Vt, vb + j0);
      tdm_load_tile((unsigned)(size_t)Dt, db + j0);
      __builtin_amdgcn_s_wait_tensorcnt(0);
    }
    __syncthreads();                       // tiles visible to all waves

    // ---- S = bias + Q*K^T  (two 16x16 tiles, fp32 C); bias gathered from global (L2-hot)
    v8f s[2];
#pragma unroll
    for (int jt = 0; jt < 2; ++jt)
#pragma unroll
      for (int r = 0; r < 8; ++r) {
        int i = i0 + r + (lh << 3);
        int j = j0 + jt * 16 + m;
        s[jt][r] = btab[ridx[i * LL + j]];
      }
#pragma unroll
    for (int kbk = 0; kbk < 8; ++kbk) {
      const __bf16* src = Qw + (kbk * 32 + lane) * 16;
      v8bf qlo = *(const v8bf*)src;
      v8bf qhi = *(const v8bf*)(src + 8);
      v16bf qa = __builtin_shufflevector(qlo, qhi, 0, 1, 2, 3, 4, 5, 6, 7,
                                         8, 9, 10, 11, 12, 13, 14, 15);
#pragma unroll
      for (int jt = 0; jt < 2; ++jt) {
        v16bf kf;                          // B 32x16: lane col N=m, K = (e&7) + 16*(e>>3) + 8*lh
#pragma unroll
        for (int e = 0; e < 16; ++e)
          kf[e] = f2bf(Kt[(kbk * 32 + (e & 7) + ((e >> 3) << 4) + (lh << 3)) * KP + jt * 16 + m]);
        s[jt] = __builtin_amdgcn_wmma_f32_16x16x32_bf16(
            false, qa, false, kf, (short)0, s[jt], false, false);
      }
    }

    // ---- online softmax update (per-row stats live in C layout)
    float sc[8];
#pragma unroll
    for (int r = 0; r < 8; ++r) {
      float t = redmax16(fmaxf(s[0][r], s[1][r]));
      float mn = fmaxf(mrow[r], t);
      sc[r] = __expf(mrow[r] - mn);
      mrow[r] = mn;
    }
#pragma unroll
    for (int jt = 0; jt < 2; ++jt)
#pragma unroll
      for (int r = 0; r < 8; ++r)
        s[jt][r] = __expf(s[jt][r] - mrow[r]);
#pragma unroll
    for (int r = 0; r < 8; ++r) {
      float rs = redsum16(s[0][r] + s[1][r]);
      lrow[r] = lrow[r] * sc[r] + rs;
    }
#pragma unroll
    for (int ct = 0; ct < 16; ++ct)
#pragma unroll
      for (int r = 0; r < 8; ++r) { accV[ct][r] *= sc[r]; accD[ct][r] *= sc[r]; }

    // ---- repack P (C layout) -> A fragment via per-wave LDS scratch
#pragma unroll
    for (int jt = 0; jt < 2; ++jt)
#pragma unroll
      for (int r = 0; r < 8; ++r)
        Ws[(r + (lh << 3)) * PSTR + jt * 16 + m] = s[jt][r];
    // A-map K runs are contiguous in chunks of 4: offsets {0,4,8,12} + 16*lh
    v16bf pa;
#pragma unroll
    for (int ch = 0; ch < 4; ++ch) {
      int koff = (ch << 2) + (lh << 4);
      f32x4 t = *(const f32x4*)(Ws + m * PSTR + koff);
#pragma unroll
      for (int e = 0; e < 4; ++e) pa[ch * 4 + e] = f2bf(t[e]);
    }

    // ---- O += P*V^T and Osd += P*std^T  (B: lane col N=c, K=j contiguous -> wide ds loads)
#pragma unroll
    for (int ct = 0; ct < 16; ++ct) {
      v16bf vf, df;
      int rowoff = (ct * 16 + m) * KP + (lh << 4);
#pragma unroll
      for (int ch = 0; ch < 4; ++ch) {
        f32x4 tv = *(const f32x4*)(Vt + rowoff + ch * 4);
        f32x4 td = *(const f32x4*)(Dt + rowoff + ch * 4);
#pragma unroll
        for (int e = 0; e < 4; ++e) {
          vf[ch * 4 + e] = f2bf(tv[e]);
          df[ch * 4 + e] = f2bf(td[e]);
        }
      }
      accV[ct] = __builtin_amdgcn_wmma_f32_16x16x32_bf16(
          false, pa, false, vf, (short)0, accV[ct], false, false);
      accD[ct] = __builtin_amdgcn_wmma_f32_16x16x32_bf16(
          false, pa, false, df, (short)0, accD[ct], false, false);
    }
  }

  // ---- epilogue: normalize, transpose tiles through LDS, coalesced [c][i] stores
  float linv[8];
#pragma unroll
  for (int r = 0; r < 8; ++r) linv[r] = 1.0f / lrow[r];

#pragma unroll
  for (int o = 0; o < 2; ++o) {
    float* outp = out + (size_t)o * ((size_t)BB * CC * LL) + (size_t)b * CC * LL + i0;
#pragma unroll
    for (int ct = 0; ct < 16; ++ct) {
      const v8f a = o ? accD[ct] : accV[ct];
#pragma unroll
      for (int r = 0; r < 8; ++r)
        Ws[m * 17 + r + (lh << 3)] = a[r] * linv[r];   // [c][i], stride 17: conflict-free
#pragma unroll
      for (int rr = 0; rr < 8; ++rr) {
        int cl = rr * 2 + lh;
        outp[(size_t)(ct * 16 + cl) * LL + m] = Ws[cl * 17 + m];
      }
    }
  }
}

extern "C" void kernel_launch(void* const* d_in, const int* in_sizes, int n_in,
                              void* d_out, int out_size, void* d_ws, size_t ws_size,
                              hipStream_t stream) {
  (void)in_sizes; (void)n_in; (void)out_size; (void)d_ws; (void)ws_size;
  const float* q  = (const float*)d_in[0];
  const float* k  = (const float*)d_in[1];
  const float* v  = (const float*)d_in[2];
  const float* sd = (const float*)d_in[3];
  const float* bt = (const float*)d_in[4];
  const int*   ri = (const int*)d_in[5];
  float* out = (float*)d_out;
  size_t smem = (size_t)(3 * CC * KP + 4 * 16 * PSTR) * sizeof(float)
              + (size_t)4 * 4096 * sizeof(__hip_bfloat16);
  attn_fa<<<dim3(BB * 16), dim3(128), smem, stream>>>(q, k, v, sd, bt, ri, out);
}
